// NeighborCooccurrenceEncoder_1889785610787
// MI455X (gfx1250) — compile-verified, matrix operand order checked by reading, not run on hardware
//
#include <hip/hip_runtime.h>

typedef __attribute__((ext_vector_type(16))) _Float16 v16h;
typedef __attribute__((ext_vector_type(8)))  float    v8f;

#define VOCAB 5000
#define NB    64
#define LL    1024
#define DD    64
#define STRIPS 8   // 16-row strips per wave

// ---------------- Phase 1: per-batch-row co-occurrence counts ----------------
// One workgroup per batch row. Two 5000-bin histograms in LDS (40 KB), built
// with ds atomics, then gathered per position. Counts written as float2
// (count_in_src, count_in_dst) for src positions [0,65536) then dst positions.
__global__ __launch_bounds__(1024) void count_kernel(const int* __restrict__ src_ids,
                                                     const int* __restrict__ dst_ids,
                                                     float2* __restrict__ cnts) {
    __shared__ unsigned int hist_s[VOCAB];
    __shared__ unsigned int hist_d[VOCAB];
    const int b = blockIdx.x;
    const int t = threadIdx.x;
    for (int i = t; i < VOCAB; i += 1024) { hist_s[i] = 0u; hist_d[i] = 0u; }
    __syncthreads();
    const int sid = src_ids[b * LL + t];
    const int did = dst_ids[b * LL + t];
    atomicAdd(&hist_s[sid], 1u);
    atomicAdd(&hist_d[did], 1u);
    __syncthreads();
    float2 cs, cd;
    cs.x = (sid == 0) ? 0.f : (float)hist_s[sid];
    cs.y = (sid == 0) ? 0.f : (float)hist_d[sid];
    cd.x = (did == 0) ? 0.f : (float)hist_s[did];
    cd.y = (did == 0) ? 0.f : (float)hist_d[did];
    cnts[b * LL + t]           = cs;
    cnts[NB * LL + b * LL + t] = cd;
}

// K index inside a 16x32 f16 A/B fragment for packed-half slot j (0..15),
// per ISA 7.12.2: VGPR0..3 -> K 2v+s (+8 for hi lanes), VGPR4..7 -> +16.
__device__ __forceinline__ int kidx(int j, int hi) {
    return ((j < 8) ? j : (j + 8)) + hi * 8;
}

// ---------------- Phase 2: fused ReLU-lift + 64x64 projection (WMMA) --------
__global__ __launch_bounds__(128) void encode_kernel(const float2* __restrict__ cnts,
                                                     const float* __restrict__ W1,
                                                     const float* __restrict__ b1,
                                                     const float* __restrict__ W2,
                                                     const float* __restrict__ b2,
                                                     float* __restrict__ out) {
    __shared__ _Float16 w2h[DD * DD];     // W2 staged as f16 (8 KB)
    __shared__ float w1s[DD], b1s[DD], b2s[DD];
    const int t = threadIdx.x;
    for (int i = t; i < DD * DD; i += 128) w2h[i] = (_Float16)W2[i];
    if (t < DD) { w1s[t] = W1[t]; b1s[t] = b1[t]; b2s[t] = b2[t]; }
    __syncthreads();

    const int wave = t >> 5;
    const int lane = t & 31;
    const int m    = lane & 15;   // row (A) / col (B,D) within tile
    const int hi   = lane >> 4;   // K-half selector for A/B fragments

    // Build the 8 B fragments (4 N-tiles x 2 K-chunks) once; B[k][n] = W2[n][k].
    v16h bf[8];
    #pragma unroll
    for (int c = 0; c < 2; ++c) {
        #pragma unroll
        for (int tn = 0; tn < 4; ++tn) {
            const int n = tn * 16 + m;
            v16h bv;
            #pragma unroll
            for (int j = 0; j < 16; ++j)
                bv[j] = w2h[n * DD + kidx(j, hi) + c * 32];
            bf[c * 4 + tn] = bv;
        }
    }
    float bias[4];
    #pragma unroll
    for (int tn = 0; tn < 4; ++tn) bias[tn] = 2.0f * b2s[tn * 16 + m];

    // Each wave: STRIPS strips of 16 rows. Total rows = 131072.
    const long baseRow = ((long)blockIdx.x * 4 + wave) * (16 * STRIPS);
    for (int s = 0; s < STRIPS; ++s) {
        const long rowBase = baseRow + (long)s * 16;
        const float2 xc = cnts[rowBase + m];   // lanes m and 16+m share row m
        v8f acc[4] = {};
        #pragma unroll
        for (int c = 0; c < 2; ++c) {
            // A fragment: g[d] = relu(x0*W1[d]+b1[d]) + relu(x1*W1[d]+b1[d])
            v16h av;
            #pragma unroll
            for (int j = 0; j < 16; ++j) {
                const int k = kidx(j, hi) + c * 32;
                const float w  = w1s[k];
                const float bb = b1s[k];
                const float g  = fmaxf(fmaf(xc.x, w, bb), 0.f)
                               + fmaxf(fmaf(xc.y, w, bb), 0.f);
                av[j] = (_Float16)g;
            }
            #pragma unroll
            for (int tn = 0; tn < 4; ++tn)
                acc[tn] = __builtin_amdgcn_wmma_f32_16x16x32_f16(
                    false, av, false, bf[c * 4 + tn], (short)0, acc[tn], false, false);
        }
        // D layout: VGPR r -> row r (lanes 0-15) / row r+8 (lanes 16-31), col = lane%16
        #pragma unroll
        for (int tn = 0; tn < 4; ++tn) {
            #pragma unroll
            for (int r = 0; r < 8; ++r) {
                out[(rowBase + r + hi * 8) * DD + tn * 16 + m] = acc[tn][r] + bias[tn];
            }
        }
    }
}

extern "C" void kernel_launch(void* const* d_in, const int* in_sizes, int n_in,
                              void* d_out, int out_size, void* d_ws, size_t ws_size,
                              hipStream_t stream) {
    (void)in_sizes; (void)n_in; (void)out_size; (void)ws_size;
    const int*   src_ids = (const int*)d_in[0];
    const int*   dst_ids = (const int*)d_in[1];
    const float* W1      = (const float*)d_in[2];
    const float* b1      = (const float*)d_in[3];
    const float* W2      = (const float*)d_in[4];
    const float* b2      = (const float*)d_in[5];
    float*       out     = (float*)d_out;
    float2*      cnts    = (float2*)d_ws;   // 131072 * 8 B = 1 MB

    count_kernel<<<NB, 1024, 0, stream>>>(src_ids, dst_ids, cnts);
    // rows = 131072, per wave 16*STRIPS = 128 rows, 4 waves/block -> 256 blocks
    encode_kernel<<<256, 128, 0, stream>>>(cnts, W1, b1, W2, b2, out);
}
